// SelfAttention_47356309405809
// MI455X (gfx1250) — matrix operand order repaired and checked by reading, not compiled, not run on hardware
//
#include <hip/hip_runtime.h>
#include <hip/hip_bf16.h>

#define EMBED 1024
#define HEADS 16
#define HDIM  64
#define NB    8
#define SEQ   1024
#define ROWS  (NB*SEQ)   // 8192

typedef _Float16 half8 __attribute__((ext_vector_type(8)));
typedef _Float16 v16h  __attribute__((ext_vector_type(16)));
typedef float    v8f   __attribute__((ext_vector_type(8)));
typedef float    f8v   __attribute__((ext_vector_type(8)));

// ---- CDNA5 async global->LDS copy path (hedged: falls back to sync loads) ----
#if defined(__has_builtin)
# if __has_builtin(__builtin_amdgcn_global_load_async_to_lds_b128) && \
     __has_builtin(__builtin_amdgcn_s_wait_asynccnt)
#  define HAVE_ASYNC_LDS 1
# endif
#endif
#ifndef HAVE_ASYNC_LDS
# define HAVE_ASYNC_LDS 0
#endif

#if HAVE_ASYNC_LDS
// Builtin wants pointers to 16-byte int vectors: global src (AS1), LDS dst (AS3).
typedef int v4i __attribute__((vector_size(16)));
typedef __attribute__((address_space(1))) v4i* gv4i_p;
typedef __attribute__((address_space(3))) v4i* lv4i_p;
#define ASYNC_CP16(gsrc, ldst) \
  __builtin_amdgcn_global_load_async_to_lds_b128((gv4i_p)(gsrc), (lv4i_p)(ldst), 0, 0)
#define WAIT_ASYNC(n) __builtin_amdgcn_s_wait_asynccnt(n)
#else
#define ASYNC_CP16(gsrc, ldst) \
  (*(half8*)(ldst) = *(const half8*)(gsrc))
#define WAIT_ASYNC(n)
#endif

__device__ __forceinline__ v8f wmma16(v16h a, v16h b, v8f c) {
  // D = A(16x32 f16) * B(32x16 f16) + C(16x16 f32)
  return __builtin_amdgcn_wmma_f32_16x16x32_f16(false, a, false, b, (short)0, c,
                                                false, false);
}

// A fragment (16x32 f16). Layout: lanes 0-15 hold K 0..15,
// lanes 16-31 hold K 16..31.  lh = lane>=16.
__device__ __forceinline__ v16h a_frag_h(const _Float16* p, int lh) {
  half8 lo = *(const half8*)(p + lh * 16);
  half8 hi = *(const half8*)(p + lh * 16 + 8);
  v16h a;
#pragma unroll
  for (int i = 0; i < 8; i++) { a[i] = lo[i]; a[8 + i] = hi[i]; }
  return a;
}

// Vectorized fp32 -> f16 conversion (8 elements / thread)
__global__ void cvt_f32_f16(const float* __restrict__ s, _Float16* __restrict__ d,
                            int n8) {
  int i = blockIdx.x * blockDim.x + threadIdx.x;
  if (i < n8) {
    f8v v = *(const f8v*)(s + (size_t)i * 8);
    half8 h;
#pragma unroll
    for (int e = 0; e < 8; e++) h[e] = (_Float16)v[e];
    *(half8*)(d + (size_t)i * 8) = h;
  }
}

// Stage one 128x32 X-tile and one 64x32 W-tile into LDS (3 x b128 per thread).
__device__ __forceinline__ void stage_tiles(const _Float16* __restrict__ Xg,
                                            const _Float16* __restrict__ Wg,
                                            _Float16* Xs, _Float16* Ws, int tid) {
  int xr = tid >> 1, xc = (tid & 1) * 16;   // X: 128 rows x 32 halves
  int wr = tid >> 2, wc = (tid & 3) * 8;    // W: 64 rows x 32 halves
  ASYNC_CP16(Xg + (size_t)xr * EMBED + xc,     Xs + xr * 32 + xc);
  ASYNC_CP16(Xg + (size_t)xr * EMBED + xc + 8, Xs + xr * 32 + xc + 8);
  ASYNC_CP16(Wg + (size_t)wr * EMBED + wc,     Ws + wr * 32 + wc);
}

// Y[r,o] = sum_k X(f16)[r,k] * W(f16)[o,k] + bias[o]
// Block: 128 rows x 64 cols, 8 waves; wave = 16 rows x 64 cols (4 C tiles).
// Double-buffered LDS staging; async prefetch of tile k+1 overlaps WMMA of k.
template <typename OutT>
__global__ __launch_bounds__(256) void gemm_ws(const _Float16* __restrict__ X,
                                               const _Float16* __restrict__ W,
                                               const float* __restrict__ bias,
                                               OutT* __restrict__ Y) {
  __shared__ _Float16 Xs[2][128 * 32];  // 8 KB each
  __shared__ _Float16 Ws[2][64 * 32];   // 4 KB each
  int tid = threadIdx.x, wave = tid >> 5, lane = tid & 31;
  int lh = lane >> 4, lc = lane & 15;
  int row0 = blockIdx.x * 128, col0 = blockIdx.y * 64;
  const _Float16* Xg = X + (size_t)row0 * EMBED;
  const _Float16* Wg = W + (size_t)col0 * EMBED;

  v8f acc[4];
#pragma unroll
  for (int t = 0; t < 4; t++) { v8f z = {}; acc[t] = z; }

  stage_tiles(Xg, Wg, Xs[0], Ws[0], tid);
  const int NT = EMBED / 32;  // 32 k-steps
  for (int kt = 0; kt < NT; ++kt) {
    int cur = kt & 1;
    if (kt + 1 < NT) {
      stage_tiles(Xg + (kt + 1) * 32, Wg + (kt + 1) * 32,
                  Xs[cur ^ 1], Ws[cur ^ 1], tid);
      WAIT_ASYNC(3);  // in-order: current tile's 3 copies done, prefetch in flight
    } else {
      WAIT_ASYNC(0);
    }
    __syncthreads();  // all waves' staging of buffer `cur` visible
    v16h a = a_frag_h(&Xs[cur][(wave * 16 + lc) * 32], lh);
#pragma unroll
    for (int t = 0; t < 4; t++) {
      v16h b = *(const v16h*)(&Ws[cur][(t * 16 + lc) * 32 + lh * 16]);
      acc[t] = wmma16(a, b, acc[t]);
    }
    __syncthreads();  // all readers done before next iteration's prefetch lands
  }

#pragma unroll
  for (int t = 0; t < 4; t++) {
    int col = col0 + t * 16 + lc;
    float bv = bias[col];
#pragma unroll
    for (int i = 0; i < 8; i++) {
      int row = row0 + wave * 16 + i + lh * 8;  // C tile: vgpr i -> M = i (+8 hi)
      Y[(size_t)row * EMBED + col] = (OutT)(acc[t][i] + bv);
    }
  }
}

// Flash-attention style: block = 128 queries of one (n,h); wave = 16 queries.
__global__ __launch_bounds__(256) void attn_kernel(const _Float16* __restrict__ Qp,
                                                   const _Float16* __restrict__ Kp,
                                                   const _Float16* __restrict__ Vp,
                                                   const int* __restrict__ mask,
                                                   _Float16* __restrict__ Op) {
  __shared__ _Float16 Ks[32 * 64];      // [key][d]      4 KB
  __shared__ _Float16 Vt[64 * 32];      // [d][key]      4 KB (transposed)
  __shared__ _Float16 Pl[8][16 * 32];   // per-wave P    8 KB

  int tid = threadIdx.x, wave = tid >> 5, lane = tid & 31;
  int lh = lane >> 4, lc = lane & 15;
  int nh = blockIdx.y;
  int n = nh / HEADS, h = nh % HEADS;
  int q0 = blockIdx.x * 128 + wave * 16;

  const _Float16* qrow = Qp + (size_t)(n * SEQ + q0 + lc) * EMBED + h * HDIM;
  v16h qa0 = a_frag_h(qrow, lh);       // d 0..31
  v16h qa1 = a_frag_h(qrow + 32, lh);  // d 32..63

  v8f o[4];
  float m[8], lsum[8];
#pragma unroll
  for (int t = 0; t < 4; t++) { v8f z = {}; o[t] = z; }
#pragma unroll
  for (int i = 0; i < 8; i++) { m[i] = -1e30f; lsum[i] = 0.0f; }

  int jload = tid >> 3;          // 0..31  (local key)
  int dload = (tid & 7) * 8;     // 0..56  (head-dim chunk)
  const float scale = 0.03125f;  // 1/sqrt(1024)

  for (int kb = 0; kb < SEQ; kb += 32) {
    __syncthreads();  // previous iteration's readers done
    {
      size_t gro = (size_t)(n * SEQ + kb + jload) * EMBED + h * HDIM + dload;
      ASYNC_CP16(Kp + gro, &Ks[jload * 64 + dload]);  // straight copy -> async
      half8 vv = *(const half8*)(Vp + gro);           // V needs transpose: manual
#pragma unroll
      for (int e = 0; e < 8; e++) Vt[(dload + e) * 32 + jload] = vv[e];
      WAIT_ASYNC(0);
    }
    __syncthreads();

    // Scores: two 16q x 16k tiles (K-dim = head dim 64 -> 2 WMMA each)
    v8f s[2];
#pragma unroll
    for (int t = 0; t < 2; t++) {
      v16h b0 = *(const v16h*)(&Ks[(t * 16 + lc) * 64 + lh * 16]);
      v16h b1 = *(const v16h*)(&Ks[(t * 16 + lc) * 64 + 32 + lh * 16]);
      v8f acc = {};
      acc = wmma16(qa0, b0, acc);
      acc = wmma16(qa1, b1, acc);
      s[t] = acc;
    }

    // Mask, scale, online-softmax stats (row = i + 8*lh, col = lane%16)
    float alpha[8];
#pragma unroll
    for (int i = 0; i < 8; i++) {
      int qr = q0 + i + lh * 8;
      const int* mr = mask + ((size_t)n * SEQ + qr) * SEQ + kb;
      float e0 = (mr[lc] == 0)      ? -3.125e18f : s[0][i] * scale;  // (-1e20)/32
      float e1 = (mr[16 + lc] == 0) ? -3.125e18f : s[1][i] * scale;
      float x = fmaxf(e0, e1);
      x = fmaxf(x, __shfl_xor(x, 1, 32));
      x = fmaxf(x, __shfl_xor(x, 2, 32));
      x = fmaxf(x, __shfl_xor(x, 4, 32));
      x = fmaxf(x, __shfl_xor(x, 8, 32));  // row max within 16-lane half
      float mn = fmaxf(m[i], x);
      alpha[i] = __expf(m[i] - mn);
      float p0 = __expf(e0 - mn);
      float p1 = __expf(e1 - mn);
      float rs = p0 + p1;
      rs += __shfl_xor(rs, 1, 32);
      rs += __shfl_xor(rs, 2, 32);
      rs += __shfl_xor(rs, 4, 32);
      rs += __shfl_xor(rs, 8, 32);
      lsum[i] = lsum[i] * alpha[i] + rs;
      m[i] = mn;
      s[0][i] = p0;
      s[1][i] = p1;
    }
#pragma unroll
    for (int t = 0; t < 4; t++)
#pragma unroll
      for (int i = 0; i < 8; i++) o[t][i] *= alpha[i];

    // Re-layout P: C-tile -> A-fragment via per-wave LDS bounce
    _Float16* Pw = &Pl[wave][0];
#pragma unroll
    for (int i = 0; i < 8; i++) {
      int r = i + lh * 8;
      Pw[r * 32 + lc]      = (_Float16)s[0][i];
      Pw[r * 32 + 16 + lc] = (_Float16)s[1][i];
    }
    __syncthreads();
    v16h pa = a_frag_h(&Pl[wave][lc * 32], lh);  // A row = lane%16, K = 32 keys
#pragma unroll
    for (int t = 0; t < 4; t++) {
      v16h vb = *(const v16h*)(&Vt[(t * 16 + lc) * 32 + lh * 16]);  // B[key, d]
      o[t] = wmma16(pa, vb, o[t]);
    }
  }

#pragma unroll
  for (int i = 0; i < 8; i++) lsum[i] = 1.0f / lsum[i];
#pragma unroll
  for (int t = 0; t < 4; t++) {
    int col = h * HDIM + t * 16 + lc;
#pragma unroll
    for (int i = 0; i < 8; i++) {
      int row = q0 + i + lh * 8;
      Op[(size_t)(n * SEQ + row) * EMBED + col] = (_Float16)(o[t][i] * lsum[i]);
    }
  }
}

extern "C" void kernel_launch(void* const* d_in, const int* in_sizes, int n_in,
                              void* d_out, int out_size, void* d_ws, size_t ws_size,
                              hipStream_t stream) {
  const float* values  = (const float*)d_in[0];
  const float* keys    = (const float*)d_in[1];
  const float* queries = (const float*)d_in[2];
  const int*   mask    = (const int*)d_in[3];
  const float* Wv = (const float*)d_in[4];
  const float* bv = (const float*)d_in[5];
  const float* Wk = (const float*)d_in[6];
  const float* bk = (const float*)d_in[7];
  const float* Wq = (const float*)d_in[8];
  const float* bq = (const float*)d_in[9];
  const float* Wo = (const float*)d_in[10];
  const float* bo = (const float*)d_in[11];

  _Float16* ws = (_Float16*)d_ws;
  const size_t RE = (size_t)ROWS * EMBED;   // 8 Mi halves
  const size_t E2 = (size_t)EMBED * EMBED;  // 1 Mi halves
  _Float16* w_v = ws;
  _Float16* w_k = w_v + E2;
  _Float16* w_q = w_k + E2;
  _Float16* w_o = w_q + E2;
  _Float16* h_v = w_o + E2;
  _Float16* h_k = h_v + RE;
  _Float16* h_q = h_k + RE;
  _Float16* p_v = h_q + RE;
  _Float16* p_k = p_v + RE;
  _Float16* p_q = p_k + RE;
  _Float16* a_o = p_q + RE;   // total: 4*E2 + 7*RE halves = 120 MB

  dim3 wgrd((unsigned)((E2 / 8 + 255) / 256));
  cvt_f32_f16<<<wgrd, 256, 0, stream>>>(Wv, w_v, (int)(E2 / 8));
  cvt_f32_f16<<<wgrd, 256, 0, stream>>>(Wk, w_k, (int)(E2 / 8));
  cvt_f32_f16<<<wgrd, 256, 0, stream>>>(Wq, w_q, (int)(E2 / 8));
  cvt_f32_f16<<<wgrd, 256, 0, stream>>>(Wo, w_o, (int)(E2 / 8));
  dim3 xgrd((unsigned)((RE / 8 + 255) / 256));
  cvt_f32_f16<<<xgrd, 256, 0, stream>>>(values,  h_v, (int)(RE / 8));
  cvt_f32_f16<<<xgrd, 256, 0, stream>>>(keys,    h_k, (int)(RE / 8));
  cvt_f32_f16<<<xgrd, 256, 0, stream>>>(queries, h_q, (int)(RE / 8));

  dim3 ggrd(ROWS / 128, EMBED / 64);  // 64 x 16
  gemm_ws<_Float16><<<ggrd, 256, 0, stream>>>(h_v, w_v, bv, p_v);
  gemm_ws<_Float16><<<ggrd, 256, 0, stream>>>(h_k, w_k, bk, p_k);
  gemm_ws<_Float16><<<ggrd, 256, 0, stream>>>(h_q, w_q, bq, p_q);

  dim3 agrd(SEQ / 128, NB * HEADS);   // 8 x 128
  attn_kernel<<<agrd, 256, 0, stream>>>(p_q, p_k, p_v, mask, a_o);

  gemm_ws<float><<<ggrd, 256, 0, stream>>>(a_o, w_o, bo, (float*)d_out);
}
